// DiscoUpBlock_81037442940979
// MI455X (gfx1250) — matrix operand order repaired and checked by reading, not compiled, not run on hardware
//
#include <hip/hip_runtime.h>
#include <hip/hip_bf16.h>
#include <stdint.h>

typedef __bf16 bf16;
typedef __attribute__((ext_vector_type(16))) __bf16 v16bf;
typedef __attribute__((ext_vector_type(8)))  __bf16 v8bf;
typedef __attribute__((ext_vector_type(8)))  float  v8f;

#define B_    2
#define CIN   128
#define HIN   91
#define WIN   180
#define NIN   (HIN*WIN)      // 16380
#define NTIL  1024           // n tiles (DISCO gemm)
#define NINP  (NTIL*16)      // 16384 (padded)
#define COUT  64
#define HOUT  181
#define WOUT  360
#define NOUT  (HOUT*WOUT)    // 65160
#define KB    9
#define NNZ   (NOUT*8)       // 521280
#define CTOT  128
#define GROUPS 8
#define NHEADS 4
#define HD    32
#define EPS   1e-5f
#define SEQ   WOUT
#define SEQP  384

__device__ __forceinline__ v8f wmma_bf16(v16bf a, v16bf b, v8f c) {
    return __builtin_amdgcn_wmma_f32_16x16x32_bf16(false, a, false, b, (short)0, c, false, false);
}
__device__ __forceinline__ v16bf cat8(v8bf lo, v8bf hh) {
    v16bf a;
    #pragma unroll
    for (int e = 0; e < 8; ++e) { a[e] = lo[e]; a[8+e] = hh[e]; }
    return a;
}
// A fragment from a contiguous 32-element bf16 row: lane half hi takes k = 16*hi .. 16*hi+15
__device__ __forceinline__ v16bf load_A_row(const bf16* row, int hi) {
    return cat8(*(const v8bf*)(row + hi*16), *(const v8bf*)(row + hi*16 + 8));
}
// B fragment from pre-packed fragment-order buffer: lane reads 32 contiguous bytes
__device__ __forceinline__ v16bf load_packed(const bf16* p, int lane) {
    const bf16* q = p + lane*16;
    return cat8(*(const v8bf*)q, *(const v8bf*)(q + 8));
}

// ---------------- packing / prep kernels ----------------
// xq packed as A-fragments: [b][nt(1024)][cs(4)][lane(32)][e(16)]
__global__ void k_pack_xq(const float* __restrict__ x, const float* __restrict__ quad,
                          bf16* __restrict__ xp, int total) {
    int i = blockIdx.x*256 + threadIdx.x;
    if (i >= total) return;
    int e = i & 15;
    int lane = (i >> 4) & 31;
    int t = i >> 9;                      // b*4096 + nt*4 + cs
    int cs = t & 3, nt = (t >> 2) & (NTIL-1), b = t >> 12;
    int hi = lane >> 4;
    int k = hi*16 + e;
    int c = cs*32 + k;
    int n = nt*16 + (lane & 15);
    bf16 v = (bf16)0.0f;
    if (n < NIN) v = (bf16)(x[((size_t)(b*CIN + c))*NIN + n] * quad[n / WIN]);
    xp[i] = v;
}

// w_disco packed as B-fragments: [kb][cs(4)][ot(4)][lane(32)][e(16)], value w[o][c][kb]
__global__ void k_pack_wdisco(const float* __restrict__ w, bf16* __restrict__ wp) {
    int i = blockIdx.x*256 + threadIdx.x;
    if (i >= KB*CIN*COUT) return;
    int e = i & 15;
    int lane = (i >> 4) & 31;
    int t = i >> 9;                      // kb*16 + cs*4 + ot
    int ot = t & 3, cs = (t >> 2) & 3, kb = t >> 4;
    int c = cs*32 + (lane >> 4)*16 + e;
    int o = ot*16 + (lane & 15);
    wp[i] = (bf16)w[(size_t)(o*CIN + c)*KB + kb];
}

// generic row-major [Kdim][Ncols] weight -> B-fragment pack [cs][nt][lane][e]
__global__ void k_pack_B(const float* __restrict__ w, bf16* __restrict__ wp,
                         int Kdim, int Ncols) {
    int i = blockIdx.x*256 + threadIdx.x;
    if (i >= Kdim*Ncols) return;
    int e = i & 15;
    int lane = (i >> 4) & 31;
    int t = i >> 9;                      // cs*(Ncols/16) + nt
    int ntiles = Ncols / 16;
    int cs = t / ntiles, nt = t % ntiles;
    int k = cs*32 + (lane >> 4)*16 + e;
    int col = nt*16 + (lane & 15);
    wp[i] = (bf16)w[(size_t)k*Ncols + col];
}

__global__ void k_fill_zero(float* __restrict__ p, int total) {
    int i = blockIdx.x*256 + threadIdx.x;
    if (i < total) p[i] = 0.0f;
}
__global__ void k_fill_zero_bf16(bf16* __restrict__ p, int total) {
    int i = blockIdx.x*256 + threadIdx.x;
    if (i < total) p[i] = (bf16)0.0f;
}

// ---------------- DISCO GEMM: xwT[b][kb][n][o] = sum_c xq[b][c][n]*w[o][c][kb] ----------------
__global__ void k_disco_gemm(const bf16* __restrict__ xp, const bf16* __restrict__ wp,
                             float* __restrict__ xwT) {
    int nt = blockIdx.x;                 // 0..1023
    int bz = blockIdx.y;                 // b*9+kb
    int kb = bz % KB;
    int lane = threadIdx.x & 31;
    int hi = lane >> 4, n = lane & 15;
    v8f z = {0,0,0,0,0,0,0,0};
    v8f a0 = z, a1 = z, a2 = z, a3 = z;
    int b = bz / KB;
    const bf16* ap = xp + ((size_t)(b*(NTIL*4) + nt*4))*512;
    const bf16* bp = wp + ((size_t)(kb*16))*512;
    #pragma unroll
    for (int cs = 0; cs < 4; ++cs) {
        v16bf a = load_packed(ap + (size_t)cs*512, lane);
        a0 = wmma_bf16(a, load_packed(bp + (size_t)(cs*4+0)*512, lane), a0);
        a1 = wmma_bf16(a, load_packed(bp + (size_t)(cs*4+1)*512, lane), a1);
        a2 = wmma_bf16(a, load_packed(bp + (size_t)(cs*4+2)*512, lane), a2);
        a3 = wmma_bf16(a, load_packed(bp + (size_t)(cs*4+3)*512, lane), a3);
    }
    float* outb = xwT + (size_t)bz*NINP*COUT;
    #pragma unroll
    for (int r = 0; r < 8; ++r) {
        float* o = outb + (size_t)(nt*16 + r + hi*8)*COUT + n;
        o[0] = a0[r]; o[16] = a1[r]; o[32] = a2[r]; o[48] = a3[r];
    }
}

// ---------------- sparse scatter-add ----------------
__global__ void k_scatter(const float* __restrict__ xwT, const float* __restrict__ psi,
                          const int* __restrict__ row_idx, const int* __restrict__ col_idx,
                          const int* __restrict__ ker_idx, float* __restrict__ y) {
    int e = blockIdx.x;
    int o = threadIdx.x;                 // 0..63
    float p = psi[e];
    int row = row_idx[e], col = col_idx[e], ker = ker_idx[e];
    #pragma unroll
    for (int b = 0; b < B_; ++b) {
        float v = p * xwT[(((size_t)(b*KB + ker))*NINP + col)*COUT + o];
        unsafeAtomicAdd(&y[((size_t)(b*NOUT + row))*COUT + o], v);
    }
}

// ---------------- GroupNorm statistics (channel-last data) ----------------
__global__ void k_gn_stats(const float* __restrict__ data, const float* __restrict__ bias,
                           float* __restrict__ meanOut, float* __restrict__ rstdOut,
                           int C, int CG) {
    int b = blockIdx.x / GROUPS, g = blockIdx.x % GROUPS;
    int tid = threadIdx.x;
    float s = 0.f, sq = 0.f;
    int total = NOUT * CG;
    const float* base = data + (size_t)b*NOUT*C + g*CG;
    for (int i = tid; i < total; i += 256) {
        int n = i / CG, c = i % CG;
        float v = base[(size_t)n*C + c];
        if (bias) v += bias[g*CG + c];
        s += v; sq += v*v;
    }
    __shared__ float ss[256], sv[256];
    ss[tid] = s; sv[tid] = sq;
    __syncthreads();
    for (int st = 128; st > 0; st >>= 1) {
        if (tid < st) { ss[tid] += ss[tid+st]; sv[tid] += sv[tid+st]; }
        __syncthreads();
    }
    if (tid == 0) {
        float mean = ss[0] / (float)total;
        float var  = sv[0] / (float)total - mean*mean;
        meanOut[blockIdx.x] = mean;
        rstdOut[blockIdx.x] = rsqrtf(var + EPS);
    }
}

// ---------------- GN1 + GELU + concat skip -> h[b][n][c] ----------------
__global__ void k_build_h(const float* __restrict__ y, const float* __restrict__ b_disco,
                          const float* __restrict__ skip, const float* __restrict__ mean,
                          const float* __restrict__ rstd, const float* __restrict__ g1,
                          const float* __restrict__ b1, float* __restrict__ h, int total) {
    int i = blockIdx.x*256 + threadIdx.x;
    if (i >= total) return;
    int c = i & 127;
    int bn = i >> 7;
    int b = bn / NOUT, n = bn % NOUT;
    float out;
    if (c < COUT) {
        int g = c >> 3;
        float v = y[(size_t)bn*COUT + c] + b_disco[c];
        v = (v - mean[b*GROUPS + g]) * rstd[b*GROUPS + g] * g1[c] + b1[c];
        out = 0.5f * v * (1.0f + erff(v * 0.70710678118654752f));
    } else {
        out = skip[((size_t)(b*COUT + (c - COUT)))*NOUT + n];
    }
    h[i] = out;
}

// ---------------- GN2 -> bf16 t ----------------
__global__ void k_gn2_apply(const float* __restrict__ h, const float* __restrict__ mean,
                            const float* __restrict__ rstd, const float* __restrict__ g2,
                            const float* __restrict__ b2, bf16* __restrict__ t, int total) {
    int i = blockIdx.x*256 + threadIdx.x;
    if (i >= total) return;
    int c = i & 127;
    int bn = i >> 7;
    int b = bn / NOUT;
    int g = c >> 4;
    t[i] = (bf16)((h[i] - mean[b*GROUPS + g]) * rstd[b*GROUPS + g] * g2[c] + b2[c]);
}

// ---------------- QKV GEMM: 16x64 per wave ----------------
__global__ void k_qkv_gemm(const bf16* __restrict__ t, const bf16* __restrict__ wp,
                           const float* __restrict__ bias, bf16* __restrict__ qkv) {
    int mt = blockIdx.x;                 // 0..8144
    int ng = blockIdx.y;                 // 0..5
    int lane = threadIdx.x & 31;
    int m = lane & 15, hi = lane >> 4, n = lane & 15;
    v8f z = {0,0,0,0,0,0,0,0};
    v8f acc[4] = {z, z, z, z};
    const bf16* arow = t + ((size_t)mt*16 + m)*CTOT;
    #pragma unroll
    for (int cs = 0; cs < 4; ++cs) {
        v16bf a = load_A_row(arow + cs*32, hi);
        #pragma unroll
        for (int tt = 0; tt < 4; ++tt)
            acc[tt] = wmma_bf16(a, load_packed(wp + (size_t)(cs*24 + ng*4 + tt)*512, lane), acc[tt]);
    }
    #pragma unroll
    for (int tt = 0; tt < 4; ++tt) {
        int col = (ng*4 + tt)*16 + n;
        float bia = bias[col];
        #pragma unroll
        for (int r = 0; r < 8; ++r) {
            int row = mt*16 + r + hi*8;
            qkv[(size_t)row*(3*CTOT) + col] = (bf16)(acc[tt][r] + bia);
        }
    }
}

// ---------------- attention: one block per (b, latitude ring, head) ----------------
__global__ void __launch_bounds__(256) k_attn(const bf16* __restrict__ qkv,
                                              float* __restrict__ obuf) {
    __shared__ bf16 Ks[SEQP][HD];        // K row-major (24 KB)
    __shared__ bf16 Vt[HD][SEQP];        // V transposed (24 KB)
    __shared__ bf16 Ps[8][16][32];       // per-wave P staging (8 KB)
    int bid = blockIdx.x;
    int b    = bid / (HOUT*NHEADS);
    int rr   = bid % (HOUT*NHEADS);
    int hrow = rr / NHEADS;
    int head = rr % NHEADS;
    int tid  = threadIdx.x;
    const bf16* rowbase = qkv + ((size_t)b*NOUT + (size_t)hrow*WOUT)*(3*CTOT);
    // zero the padded tokens
    for (int i = tid; i < (SEQP-SEQ)*HD; i += 256) {
        int tok = SEQ + i / HD, d = i % HD;
        Ks[tok][d] = (bf16)0.0f;
        Vt[d][tok] = (bf16)0.0f;
    }
    // async-copy K rows (4 x b128 per token) -- CDNA5 async global->LDS path
    for (int i = tid; i < SEQ*4; i += 256) {
        int tok = i >> 2, part = i & 3;
        const bf16* gk = rowbase + (size_t)tok*(3*CTOT) + CTOT + head*HD + part*8;
        unsigned lk = (unsigned)(uintptr_t)&Ks[tok][part*8];
        asm volatile("global_load_async_to_lds_b128 %0, %1, off" :: "v"(lk), "v"(gk) : "memory");
    }
    // V staged transposed (regular loads, scattered ds stores)
    for (int i = tid; i < SEQ*HD; i += 256) {
        int tok = i >> 5, d = i & 31;
        Vt[d][tok] = rowbase[(size_t)tok*(3*CTOT) + 2*CTOT + head*HD + d];
    }
    asm volatile("s_wait_asynccnt 0" ::: "memory");
    __syncthreads();
    int wave = tid >> 5, lane = tid & 31;
    int m = lane & 15, hi = lane >> 4, n = lane & 15;
    const float sc = 0.17677669529663687f;  // 1/sqrt(32)
    for (int qt = wave; qt < 23; qt += 8) {
        const bf16* qrow = rowbase + (size_t)(qt*16 + m)*(3*CTOT) + head*HD;
        v16bf aq = load_A_row(qrow, hi);    // rows >=360 read padded garbage; never stored
        float mrow[8], lrow[8];
        #pragma unroll
        for (int r = 0; r < 8; ++r) { mrow[r] = -3.0e38f; lrow[r] = 0.0f; }
        v8f o0 = {0,0,0,0,0,0,0,0}, o1 = {0,0,0,0,0,0,0,0};
        for (int j = 0; j < SEQP/32; ++j) {
            // K fragments: contiguous per-lane 32B from LDS
            v16bf bk0 = cat8(*(const v8bf*)&Ks[j*32 + n][hi*16],
                             *(const v8bf*)&Ks[j*32 + n][hi*16 + 8]);
            v16bf bk1 = cat8(*(const v8bf*)&Ks[j*32 + 16 + n][hi*16],
                             *(const v8bf*)&Ks[j*32 + 16 + n][hi*16 + 8]);
            v8f z = {0,0,0,0,0,0,0,0};
            v8f s0 = wmma_bf16(aq, bk0, z);
            v8f s1 = wmma_bf16(aq, bk1, z);
            float bias0 = (j*32 + n)      < SEQ ? 0.0f : -1.0e30f;
            float bias1 = (j*32 + 16 + n) < SEQ ? 0.0f : -1.0e30f;
            float rm[8];
            #pragma unroll
            for (int r = 0; r < 8; ++r) {
                s0[r] = s0[r]*sc + bias0;
                s1[r] = s1[r]*sc + bias1;
                rm[r] = fmaxf(s0[r], s1[r]);
            }
            #pragma unroll
            for (int off = 1; off < 16; off <<= 1) {
                #pragma unroll
                for (int r = 0; r < 8; ++r) rm[r] = fmaxf(rm[r], __shfl_xor(rm[r], off, 32));
            }
            float p0[8], p1[8], rs[8];
            #pragma unroll
            for (int r = 0; r < 8; ++r) {
                float mn = fmaxf(mrow[r], rm[r]);
                float scal = __expf(mrow[r] - mn);
                p0[r] = __expf(s0[r] - mn);
                p1[r] = __expf(s1[r] - mn);
                rs[r] = p0[r] + p1[r];
                mrow[r] = mn;
                lrow[r] *= scal;
                o0[r] *= scal; o1[r] *= scal;
            }
            #pragma unroll
            for (int off = 1; off < 16; off <<= 1) {
                #pragma unroll
                for (int r = 0; r < 8; ++r) rs[r] += __shfl_xor(rs[r], off, 32);
            }
            #pragma unroll
            for (int r = 0; r < 8; ++r) lrow[r] += rs[r];
            // C-layout P -> LDS -> A-layout P
            #pragma unroll
            for (int r = 0; r < 8; ++r) {
                Ps[wave][r + hi*8][n]      = (bf16)p0[r];
                Ps[wave][r + hi*8][16 + n] = (bf16)p1[r];
            }
            asm volatile("s_wait_dscnt 0" ::: "memory");
            v16bf pa = load_A_row(&Ps[wave][m][0], hi);
            // V fragments from transposed LDS: contiguous per-lane 32B
            v16bf bv0 = cat8(*(const v8bf*)&Vt[n][j*32 + hi*16],
                             *(const v8bf*)&Vt[n][j*32 + hi*16 + 8]);
            v16bf bv1 = cat8(*(const v8bf*)&Vt[16 + n][j*32 + hi*16],
                             *(const v8bf*)&Vt[16 + n][j*32 + hi*16 + 8]);
            o0 = wmma_bf16(pa, bv0, o0);
            o1 = wmma_bf16(pa, bv1, o1);
        }
        float* dst = obuf + ((size_t)b*NOUT + (size_t)hrow*WOUT)*CTOT + head*HD;
        #pragma unroll
        for (int r = 0; r < 8; ++r) {
            int q = qt*16 + r + hi*8;
            if (q < SEQ) {
                float inv = 1.0f / lrow[r];
                dst[(size_t)q*CTOT + n]      = o0[r]*inv;
                dst[(size_t)q*CTOT + 16 + n] = o1[r]*inv;
            }
        }
    }
}

// ---------------- output projection + bias + residual (16x64 per wave) ----------------
__global__ void k_proj_gemm(const float* __restrict__ obuf, const bf16* __restrict__ wp,
                            const float* __restrict__ bo, const float* __restrict__ h,
                            float* __restrict__ out) {
    int mt = blockIdx.x;                 // 0..8144
    int ng = blockIdx.y;                 // 0..1
    int lane = threadIdx.x & 31;
    int m = lane & 15, hi = lane >> 4, n = lane & 15;
    v8f z = {0,0,0,0,0,0,0,0};
    v8f acc[4] = {z, z, z, z};
    const float* arow = obuf + ((size_t)mt*16 + m)*CTOT;
    #pragma unroll
    for (int cs = 0; cs < 4; ++cs) {
        v8f lo = *(const v8f*)(arow + cs*32 + hi*16);
        v8f hh = *(const v8f*)(arow + cs*32 + hi*16 + 8);
        v16bf a;
        #pragma unroll
        for (int e = 0; e < 8; ++e) { a[e] = (bf16)lo[e]; a[8+e] = (bf16)hh[e]; }
        #pragma unroll
        for (int tt = 0; tt < 4; ++tt)
            acc[tt] = wmma_bf16(a, load_packed(wp + (size_t)(cs*8 + ng*4 + tt)*512, lane), acc[tt]);
    }
    #pragma unroll
    for (int tt = 0; tt < 4; ++tt) {
        int col = (ng*4 + tt)*16 + n;
        float bia = bo[col];
        #pragma unroll
        for (int r = 0; r < 8; ++r) {
            int row = mt*16 + r + hi*8;
            int bb = row / NOUT, sp = row % NOUT;
            out[((size_t)bb*CTOT + col)*NOUT + sp] = acc[tt][r] + bia + h[(size_t)row*CTOT + col];
        }
    }
}

extern "C" void kernel_launch(void* const* d_in, const int* in_sizes, int n_in,
                              void* d_out, int out_size, void* d_ws, size_t ws_size,
                              hipStream_t stream) {
    const float* x       = (const float*)d_in[0];
    const float* skip    = (const float*)d_in[1];
    const float* w_disco = (const float*)d_in[2];
    const float* b_disco = (const float*)d_in[3];
    const float* psi     = (const float*)d_in[4];
    const float* quad    = (const float*)d_in[5];
    const float* g1      = (const float*)d_in[6];
    const float* b1      = (const float*)d_in[7];
    const float* g2      = (const float*)d_in[8];
    const float* b2      = (const float*)d_in[9];
    const float* wqkv    = (const float*)d_in[10];
    const float* bqkv    = (const float*)d_in[11];
    const float* wo      = (const float*)d_in[12];
    const float* bo      = (const float*)d_in[13];
    const int* row_idx   = (const int*)d_in[14];
    const int* col_idx   = (const int*)d_in[15];
    const int* ker_idx   = (const int*)d_in[16];
    float* out = (float*)d_out;
    (void)in_sizes; (void)n_in; (void)out_size; (void)ws_size;

    char* wsp = (char*)d_ws;
    size_t off = 0;
    auto alloc = [&](size_t bytes) -> void* {
        void* p = wsp + off;
        off += (bytes + 255) & ~(size_t)255;
        return p;
    };
    const int totXqp = B_*NTIL*4*512;                        // packed xq elements
    bf16*  xqp     = (bf16*) alloc((size_t)totXqp*2);
    bf16*  wkp     = (bf16*) alloc((size_t)KB*CIN*COUT*2);
    bf16*  wqkvp   = (bf16*) alloc((size_t)CTOT*3*CTOT*2);
    bf16*  wop     = (bf16*) alloc((size_t)CTOT*CTOT*2);
    float* xwT     = (float*)alloc((size_t)B_*KB*NINP*COUT*4);
    float* y       = (float*)alloc((size_t)B_*NOUT*COUT*4);
    float* h       = (float*)alloc((size_t)B_*NOUT*CTOT*4);
    bf16*  t_bf    = (bf16*) alloc((size_t)B_*NOUT*CTOT*2);
    bf16*  qkv_bf  = (bf16*) alloc(((size_t)B_*NOUT + 16)*3*CTOT*2);  // +16 pad rows
    float* obuf    = (float*)alloc((size_t)B_*NOUT*CTOT*4);
    float* gn1m    = (float*)alloc(64);
    float* gn1r    = (float*)alloc(64);
    float* gn2m    = (float*)alloc(64);
    float* gn2r    = (float*)alloc(64);

    k_pack_xq<<<(totXqp+255)/256, 256, 0, stream>>>(x, quad, xqp, totXqp);
    k_pack_wdisco<<<(KB*CIN*COUT+255)/256, 256, 0, stream>>>(w_disco, wkp);
    k_pack_B<<<(CTOT*3*CTOT+255)/256, 256, 0, stream>>>(wqkv, wqkvp, CTOT, 3*CTOT);
    k_pack_B<<<(CTOT*CTOT+255)/256, 256, 0, stream>>>(wo, wop, CTOT, CTOT);
    k_fill_zero<<<(B_*NOUT*COUT+255)/256, 256, 0, stream>>>(y, B_*NOUT*COUT);
    k_fill_zero_bf16<<<(16*3*CTOT+255)/256, 256, 0, stream>>>(qkv_bf + (size_t)B_*NOUT*3*CTOT, 16*3*CTOT);

    dim3 gDisco(NTIL, B_*KB);
    k_disco_gemm<<<gDisco, 32, 0, stream>>>(xqp, wkp, xwT);

    k_scatter<<<NNZ, COUT, 0, stream>>>(xwT, psi, row_idx, col_idx, ker_idx, y);

    k_gn_stats<<<B_*GROUPS, 256, 0, stream>>>(y, b_disco, gn1m, gn1r, COUT, COUT/GROUPS);
    int totH = B_*NOUT*CTOT;
    k_build_h<<<(totH+255)/256, 256, 0, stream>>>(y, b_disco, skip, gn1m, gn1r, g1, b1, h, totH);
    k_gn_stats<<<B_*GROUPS, 256, 0, stream>>>(h, nullptr, gn2m, gn2r, CTOT, CTOT/GROUPS);
    k_gn2_apply<<<(totH+255)/256, 256, 0, stream>>>(h, gn2m, gn2r, g2, b2, t_bf, totH);

    dim3 gQkv(B_*NOUT/16, 6);
    k_qkv_gemm<<<gQkv, 32, 0, stream>>>(t_bf, wqkvp, bqkv, qkv_bf);

    k_attn<<<B_*HOUT*NHEADS, 256, 0, stream>>>(qkv_bf, obuf);

    dim3 gProj(B_*NOUT/16, 2);
    k_proj_gemm<<<gProj, 32, 0, stream>>>(obuf, wop, bo, h, out);
}
